// Interpolate1D_54288386622106
// MI455X (gfx1250) — compile-verified, matrix operand order checked by reading, not run on hardware
//
#include <hip/hip_runtime.h>
#include <hip/hip_bf16.h>
#include <math.h>

typedef __attribute__((ext_vector_type(16))) _Float16 v16h;
typedef __attribute__((ext_vector_type(8)))  _Float16 v8h;
typedef __attribute__((ext_vector_type(8)))  float    v8f;

#define D_DIM 64
#define R_DIM 256
#define ROWS_PER_WAVE 16
#define WAVES_PER_BLOCK 4
#define BLOCK_THREADS 128
#define ROWS_PER_BLOCK 64
#define FSTRIDE 264   // 256 + 8 pad floats; keeps per-row chunks 32B-aligned

__global__ __launch_bounds__(BLOCK_THREADS)
void interp1d_fused_kernel(const float* __restrict__ y,
                           const float* __restrict__ x,
                           const float* __restrict__ W,
                           const float* __restrict__ bvec,
                           const float* __restrict__ logdet_in,
                           const float* __restrict__ bp,
                           float* __restrict__ out_z,
                           float* __restrict__ out_x,
                           float* __restrict__ out_ld)
{
    __shared__ __align__(16) _Float16 sWt[R_DIM * D_DIM];                 // W transposed, f16 (32 KB)
    __shared__ __align__(16) float sFeat[WAVES_PER_BLOCK * ROWS_PER_WAVE * FSTRIDE]; // 66 KB
    __shared__ float sBP[R_DIM];
    __shared__ float sBias[R_DIM];

    const int tid  = threadIdx.x;
    const int lane = tid & 31;
    const int wave = tid >> 5;
    const int blockRow0 = blockIdx.x * ROWS_PER_BLOCK;

    // ---- preload W (transposed, converted to f16), bias, base_points ----
    for (int i = tid; i < (D_DIM * R_DIM) / 4; i += BLOCK_THREADS) {
        float4 wv = ((const float4*)W)[i];
        int k = (i * 4) / R_DIM;
        int n = (i * 4) % R_DIM;
        sWt[(n + 0) * D_DIM + k] = (_Float16)wv.x;
        sWt[(n + 1) * D_DIM + k] = (_Float16)wv.y;
        sWt[(n + 2) * D_DIM + k] = (_Float16)wv.z;
        sWt[(n + 3) * D_DIM + k] = (_Float16)wv.w;
    }
    for (int i = tid; i < R_DIM; i += BLOCK_THREADS) {
        sBP[i]   = bp[i];
        sBias[i] = bvec[i];
    }

    // ---- x passthrough to output (pure bandwidth, coalesced b128) ----
    {
        const float4* src = (const float4*)(x + (size_t)blockRow0 * D_DIM);
        float4*       dst = (float4*)(out_x + (size_t)blockRow0 * D_DIM);
        #pragma unroll 2
        for (int i = tid; i < ROWS_PER_BLOCK * D_DIM / 4; i += BLOCK_THREADS)
            dst[i] = src[i];
    }
    __syncthreads();

    // ---- per-wave GEMM: 16 rows x 256 cols, K = 64, v_wmma_f32_16x16x32_f16 ----
    const int m0 = blockRow0 + wave * ROWS_PER_WAVE;
    const int r  = lane & 15;    // row (A) / column (B) within 16-wide tile
    const int hb = lane >> 4;    // half-wave K split per ISA fragment layout
    float* feat = &sFeat[wave * ROWS_PER_WAVE * FSTRIDE];

    const float* xr = x + (size_t)(m0 + r) * D_DIM;

    // A fragments (16x32 f16 each): lane holds K-octets {hb*8 .. hb*8+7, 16+hb*8 .. 16+hb*8+7}
    v8f a00 = *(const v8f*)(xr +      hb * 8);
    v8f a01 = *(const v8f*)(xr + 16 + hb * 8);
    v8f a10 = *(const v8f*)(xr + 32 + hb * 8);
    v8f a11 = *(const v8f*)(xr + 48 + hb * 8);
    v16h afrag0 = __builtin_shufflevector(__builtin_convertvector(a00, v8h),
                                          __builtin_convertvector(a01, v8h),
                                          0,1,2,3,4,5,6,7,8,9,10,11,12,13,14,15);
    v16h afrag1 = __builtin_shufflevector(__builtin_convertvector(a10, v8h),
                                          __builtin_convertvector(a11, v8h),
                                          0,1,2,3,4,5,6,7,8,9,10,11,12,13,14,15);

    #pragma unroll 4
    for (int t = 0; t < R_DIM / 16; ++t) {
        const int n = t * 16 + r;
        const _Float16* wr = &sWt[n * D_DIM];
        v8h b00 = *(const v8h*)(wr +      hb * 8);
        v8h b01 = *(const v8h*)(wr + 16 + hb * 8);
        v8h b10 = *(const v8h*)(wr + 32 + hb * 8);
        v8h b11 = *(const v8h*)(wr + 48 + hb * 8);
        v16h bfrag0 = __builtin_shufflevector(b00, b01, 0,1,2,3,4,5,6,7,8,9,10,11,12,13,14,15);
        v16h bfrag1 = __builtin_shufflevector(b10, b11, 0,1,2,3,4,5,6,7,8,9,10,11,12,13,14,15);

        v8f c = {};
        c = __builtin_amdgcn_wmma_f32_16x16x32_f16(false, afrag0, false, bfrag0,
                                                   (short)0, c, false, false);
        c = __builtin_amdgcn_wmma_f32_16x16x32_f16(false, afrag1, false, bfrag1,
                                                   (short)0, c, false, false);

        // C layout: VGPR v -> row v + 8*hb, column = lane&15 within tile
        #pragma unroll
        for (int v = 0; v < 8; ++v)
            feat[(v + hb * 8) * FSTRIDE + t * 16 + r] = c[v];
    }

    // ---- softmax + cumsum per row (wave-cooperative, 8 cols/lane) ----
    for (int rr = 0; rr < ROWS_PER_WAVE; ++rr) {
        float* row = feat + rr * FSTRIDE;
        const int c0 = lane * 8;
        float v[8];
        #pragma unroll
        for (int j = 0; j < 8; ++j) v[j] = row[c0 + j] + sBias[c0 + j];

        float lmax = v[0];
        #pragma unroll
        for (int j = 1; j < 8; ++j) lmax = fmaxf(lmax, v[j]);
        #pragma unroll
        for (int m = 16; m >= 1; m >>= 1) lmax = fmaxf(lmax, __shfl_xor(lmax, m, 32));

        float pre[8];
        float lsum = 0.f;
        #pragma unroll
        for (int j = 0; j < 8; ++j) { float e = __expf(v[j] - lmax); lsum += e; pre[j] = lsum; }

        float scan = lsum;
        #pragma unroll
        for (int d = 1; d < 32; d <<= 1) {
            float t2 = __shfl_up(scan, d, 32);
            if (lane >= d) scan += t2;
        }
        float total = __shfl(scan, 31, 32);
        float excl  = scan - lsum;
        float inv   = 1.0f / total;
        #pragma unroll
        for (int j = 0; j < 8; ++j) row[c0 + j] = (excl + pre[j]) * inv;
    }

    // ---- piecewise-linear interpolation at y, one row per lane (lanes 0..15) ----
    if (lane < ROWS_PER_WAVE) {
        const int grow = m0 + lane;
        const float* row = feat + lane * FSTRIDE;
        const float yy = y[grow];
        int start = (int)floorf(yy * (float)(R_DIM - 1));
        start = start < 0 ? 0 : (start > R_DIM - 2 ? R_DIM - 2 : start);
        // exact searchsorted(side='right')-1 against the actual base_points values
        while (start < R_DIM - 2 && sBP[start + 1] <= yy) ++start;
        while (start > 0 && sBP[start] > yy) --start;
        const float x0 = sBP[start], x1 = sBP[start + 1];
        const float f0 = row[start], f1 = row[start + 1];
        const float slope = (f1 - f0) / (x1 - x0);
        out_z[grow]  = f0 + slope * (yy - x0);   // SCALE=1, OFFSET=0
        out_ld[grow] = logdet_in[grow] + __logf(fabsf(slope));
    }
}

extern "C" void kernel_launch(void* const* d_in, const int* in_sizes, int n_in,
                              void* d_out, int out_size, void* d_ws, size_t ws_size,
                              hipStream_t stream) {
    // setup_inputs order: y, x, W, b, logdet, base_points
    const float* y  = (const float*)d_in[0];
    const float* x  = (const float*)d_in[1];
    const float* W  = (const float*)d_in[2];
    const float* b  = (const float*)d_in[3];
    const float* ld = (const float*)d_in[4];
    const float* bp = (const float*)d_in[5];

    const int nB = in_sizes[0];             // B = 524288
    float* out    = (float*)d_out;          // [z (B)] [x (B*64)] [logdet (B)]
    float* out_z  = out;
    float* out_x  = out + nB;
    float* out_ld = out + nB + (size_t)nB * D_DIM;

    const int grid = nB / ROWS_PER_BLOCK;   // 8192 blocks of 128 threads (4 waves)
    interp1d_fused_kernel<<<grid, BLOCK_THREADS, 0, stream>>>(
        y, x, W, b, ld, bp, out_z, out_x, out_ld);
}
